// SO2_Convolution_55516747268851
// MI455X (gfx1250) — matrix operand order repaired and checked by reading, not compiled, not run on hardware
//
#include <hip/hip_runtime.h>

typedef __bf16 v16bf __attribute__((ext_vector_type(16)));
typedef float  v8f   __attribute__((ext_vector_type(8)));

// PERM (compile-time from reference): lp->mp permutation of the 29 coefficients
__constant__ int PERM_D[29] = {0,2,6,11,16,21,26, 3,7,12,17,22,27,
                               1,5,10,15,20,25, 8,13,18,23,28, 4,9,14,19,24};

__device__ __forceinline__ unsigned short f2bf(float f) {
  unsigned u = __float_as_uint(f);
  return (unsigned short)((u + 0x7FFFu + ((u >> 16) & 1u)) >> 16);  // RNE
}
__device__ __forceinline__ float bflo(unsigned u) { return __uint_as_float(u << 16); }
__device__ __forceinline__ float bfhi(unsigned u) { return __uint_as_float(u & 0xFFFF0000u); }
__device__ __forceinline__ unsigned pk2(float a, float b) {
  return (unsigned)f2bf(a) | ((unsigned)f2bf(b) << 16);
}

union AFrag { v16bf v; uint4 q[2]; };

// ---------------------------------------------------------------------------
// Weight prep: fp32 -> bf16, with complex-recombination folding for m1/m2.
// ---------------------------------------------------------------------------
__global__ __launch_bounds__(256) void so2_prep_weights(
    const float* __restrict__ w2, const float* __restrict__ wm0,
    const float* __restrict__ wm1, const float* __restrict__ wm2,
    unsigned short* __restrict__ W2bf, unsigned short* __restrict__ Wm0bf,
    unsigned short* __restrict__ Wm1bf, unsigned short* __restrict__ Wm2bf) {
  const long s0 = 2304L * 128, s1 = 896L * 896, s2 = 1536L * 1536, s3 = 1280L * 1280;
  long idx = (long)blockIdx.x * 256 + threadIdx.x;
  if (idx >= s0 + s1 + s2 + s3) return;
  if (idx < s0) { W2bf[idx] = f2bf(w2[idx]); return; }
  idx -= s0;
  if (idx < s1) { Wm0bf[idx] = f2bf(wm0[idx]); return; }
  idx -= s1;
  if (idx < s2) {
    long n = idx / 1536, k = idx % 1536;
    float v;
    if (k < 768) v = wm1[n * 768 + k];
    else {
      float sg = (n < 768) ? -1.f : 1.f;
      long n2 = (n + 768) % 1536;
      v = sg * wm1[n2 * 768 + (k - 768)];
    }
    Wm1bf[idx] = f2bf(v);
    return;
  }
  idx -= s2;
  {
    long n = idx / 1280, k = idx % 1280;
    float v;
    if (k < 640) v = wm2[n * 640 + k];
    else {
      float sg = (n < 640) ? -1.f : 1.f;
      long n2 = (n + 640) % 1280;
      v = sg * wm2[n2 * 640 + (k - 640)];
    }
    Wm2bf[idx] = f2bf(v);
  }
}

// ---------------------------------------------------------------------------
// h = silu(LN(x_edge @ W1^T + b1)) -> bf16.  16 edges / block, 256 threads.
// ---------------------------------------------------------------------------
__global__ __launch_bounds__(256) void so2_h_kernel(
    const float* __restrict__ x_edge, const float* __restrict__ w1,
    const float* __restrict__ b1, const float* __restrict__ g,
    const float* __restrict__ beta, unsigned short* __restrict__ h_out, int E) {
  __shared__ unsigned short W1s[128 * 130];
  __shared__ float xs[16][128];
  __shared__ float hs[16][128];
  const int t = threadIdx.x;
  const long e0 = (long)blockIdx.x * 16;

  for (int i = t; i < 128 * 128; i += 256) {
    int r = i >> 7, c = i & 127;
    W1s[r * 130 + c] = f2bf(w1[i]);
  }
  for (int i = t; i < 16 * 128; i += 256) {
    int r = i >> 7, c = i & 127;
    long e = e0 + r;
    xs[r][c] = (e < E) ? x_edge[e * 128 + c] : 0.f;
  }
  __syncthreads();

  const int j = t & 127;
  const int esub = t >> 7;
  float acc[8] = {0.f, 0.f, 0.f, 0.f, 0.f, 0.f, 0.f, 0.f};
  for (int k = 0; k < 128; ++k) {
    const float wv = bflo((unsigned)W1s[j * 130 + k]);
#pragma unroll
    for (int p = 0; p < 8; ++p) acc[p] += wv * xs[p * 2 + esub][k];
  }
  const float bj = b1[j];
#pragma unroll
  for (int p = 0; p < 8; ++p) hs[p * 2 + esub][j] = acc[p] + bj;
  __syncthreads();

  const int wvid = t >> 5;
  const int lane = t & 31;
  for (int ee = 0; ee < 2; ++ee) {
    const int e = wvid * 2 + ee;
    float s = 0.f, s2 = 0.f;
#pragma unroll
    for (int q = 0; q < 4; ++q) {
      float v = hs[e][lane + 32 * q];
      s += v;
      s2 += v * v;
    }
#pragma unroll
    for (int m = 16; m >= 1; m >>= 1) {
      s += __shfl_xor(s, m, 32);
      s2 += __shfl_xor(s2, m, 32);
    }
    const float mu = s * (1.f / 128.f);
    const float var = s2 * (1.f / 128.f) - mu * mu;
    const float inv = rsqrtf(var + 1e-5f);
    if (e0 + e < E) {
#pragma unroll
      for (int q = 0; q < 4; ++q) {
        const int jj = lane + 32 * q;
        float y = (hs[e][jj] - mu) * inv * g[jj] + beta[jj];
        y = y / (1.f + __expf(-y));  // silu
        h_out[(e0 + e) * 128 + jj] = f2bf(y);
      }
    }
  }
}

// ---------------------------------------------------------------------------
// Generic bf16 WMMA GEMM:  out[e, n0+n] = sum_k A[e,k] * W[n][k] (+bias[n])
// BM=128, BN=128, BK=32; 256 threads = 8 waves (4 in M x 2 in N),
// each wave: 2 M-tiles x 4 N-tiles of v_wmma_f32_16x16x32_bf16.
// Double-buffered LDS; B (and mode-0 A) staged with CDNA5
// global_load_async_to_lds_b128 (ASYNCcnt), one barrier per K-step.
// ---------------------------------------------------------------------------
__global__ __launch_bounds__(256) void so2_wmma_gemm(
    const unsigned short* __restrict__ Abf, int lda,
    const float* __restrict__ xemb, const unsigned short* __restrict__ rad, int gid,
    const unsigned short* __restrict__ W,
    const float* __restrict__ bias,
    void* __restrict__ outp, int ldo, int outBf16,
    int E, int N, int K) {
  __shared__ __align__(16) unsigned short At[2][128 * 32];  // 2 x 8KB
  __shared__ __align__(16) unsigned short Bt[2][128 * 32];  // 2 x 8KB

  const int t = threadIdx.x;
  const long e0 = (long)blockIdx.y * 128;
  const long n0 = (long)blockIdx.x * 128;

  const int srow = t >> 1;   // staging row 0..127
  const int shalf = t & 1;   // staging 16-element half

  const int lane = t & 31;
  const int wv = t >> 5;
  const int mi = wv >> 1;    // 0..3 (M sub-block of 32 rows)
  const int ni = wv & 1;     // 0..1 (N sub-block of 64 cols)
  const int lr = lane & 15;
  const int lh = lane >> 4;

  const long e = e0 + srow;

  auto stage = [&](int b, int k0) {
    // ---- B tile: Bt[b][n][k] = W[n0+n][k0+k]; async DMA global->LDS, 32B/thread
    {
      const unsigned short* gsrc = W + (size_t)(n0 + srow) * K + k0 + shalf * 16;
      unsigned ldsb = (unsigned)(size_t)&Bt[b][srow * 32 + shalf * 16];
      asm volatile(
          "global_load_async_to_lds_b128 %0, %1, off\n\t"
          "global_load_async_to_lds_b128 %0, %1, off offset:16"
          :: "v"(ldsb), "v"(gsrc) : "memory");
    }
    // ---- A tile
    if (xemb) {  // mode 1: A[e,k] = x_emb[e, PERM-row] * rad  (VALU convert + ds_store)
      const int r = k0 >> 7;
      int embrow, radrow;
      if (gid == 0)      { embrow = PERM_D[r];      radrow = r; }
      else if (gid == 1) { embrow = PERM_D[7 + r];  radrow = 7 + (r % 6); }
      else               { embrow = PERM_D[19 + r]; radrow = 13 + (r % 5); }
      const int cb = (k0 & 127) + shalf * 16;
      uint4 o0 = make_uint4(0, 0, 0, 0), o1 = make_uint4(0, 0, 0, 0);
      if (e < E) {
        const float4* xv = (const float4*)(xemb + ((e * 29 + embrow) << 7) + cb);
        const uint4* rv = (const uint4*)(rad + e * 2304 + radrow * 128 + cb);
        float4 xa = xv[0], xb = xv[1], xc = xv[2], xd = xv[3];
        uint4 ra = rv[0], rb = rv[1];
        o0.x = pk2(xa.x * bflo(ra.x), xa.y * bfhi(ra.x));
        o0.y = pk2(xa.z * bflo(ra.y), xa.w * bfhi(ra.y));
        o0.z = pk2(xb.x * bflo(ra.z), xb.y * bfhi(ra.z));
        o0.w = pk2(xb.z * bflo(ra.w), xb.w * bfhi(ra.w));
        o1.x = pk2(xc.x * bflo(rb.x), xc.y * bfhi(rb.x));
        o1.y = pk2(xc.z * bflo(rb.y), xc.w * bfhi(rb.y));
        o1.z = pk2(xd.x * bflo(rb.z), xd.y * bfhi(rb.z));
        o1.w = pk2(xd.z * bflo(rb.w), xd.w * bfhi(rb.w));
      }
      *(uint4*)(&At[b][srow * 32 + shalf * 16]) = o0;
      *(uint4*)(&At[b][srow * 32 + shalf * 16 + 8]) = o1;
    } else {  // mode 0: plain bf16 copy -> async DMA (stray tail rows land in
              // the adjacent ws region and only feed guarded-out output rows)
      const unsigned short* gsrc = Abf + (size_t)e * lda + k0 + shalf * 16;
      unsigned ldsa = (unsigned)(size_t)&At[b][srow * 32 + shalf * 16];
      asm volatile(
          "global_load_async_to_lds_b128 %0, %1, off\n\t"
          "global_load_async_to_lds_b128 %0, %1, off offset:16"
          :: "v"(ldsa), "v"(gsrc) : "memory");
    }
  };

  v8f acc[2][4];
  const v8f vz = {0.f, 0.f, 0.f, 0.f, 0.f, 0.f, 0.f, 0.f};
#pragma unroll
  for (int mt = 0; mt < 2; ++mt)
#pragma unroll
    for (int nt = 0; nt < 4; ++nt) acc[mt][nt] = vz;

  // prologue: fill buffer 0
  stage(0, 0);
  asm volatile("s_wait_asynccnt 0x0" ::: "memory");
  __syncthreads();

  int buf = 0;
  for (int k0 = 0; k0 < K; k0 += 32) {
    const int nxt = k0 + 32;
    if (nxt < K) stage(buf ^ 1, nxt);  // prefetch next tile into other buffer

    // ---- fragments (contiguous k order, same in A and B) + WMMA on current buffer
    AFrag fa[2], fb[4];
#pragma unroll
    for (int mt = 0; mt < 2; ++mt) {
      // A 16x32: lanes<16 hold K 0..15; lanes>=16 hold K 16..31
      const unsigned short* p = &At[buf][(mi * 32 + mt * 16 + lr) * 32];
      fa[mt].q[0] = *(const uint4*)(p + lh * 16);
      fa[mt].q[1] = *(const uint4*)(p + lh * 16 + 8);
    }
#pragma unroll
    for (int nt = 0; nt < 4; ++nt) {
      // B 32x16: lane = khalf*16 + n; per lane 16 contiguous K of column n
      const unsigned short* p = &Bt[buf][(ni * 64 + nt * 16 + lr) * 32 + lh * 16];
      fb[nt].q[0] = *(const uint4*)(p);
      fb[nt].q[1] = *(const uint4*)(p + 8);
    }
#pragma unroll
    for (int mt = 0; mt < 2; ++mt)
#pragma unroll
      for (int nt = 0; nt < 4; ++nt)
        acc[mt][nt] = __builtin_amdgcn_wmma_f32_16x16x32_bf16(
            false, fa[mt].v, false, fb[nt].v, (short)0, acc[mt][nt], false, false);

    if (nxt < K) {
      asm volatile("s_wait_asynccnt 0x0" ::: "memory");
      __syncthreads();
    }
    buf ^= 1;
  }

  // ---- epilogue: C/D layout: VGPR i -> M = i + 8*(lane>=16), N = lane&15
#pragma unroll
  for (int mt = 0; mt < 2; ++mt) {
#pragma unroll
    for (int nt = 0; nt < 4; ++nt) {
      const long colg = n0 + ni * 64 + nt * 16 + lr;
      const float bv = bias ? bias[colg] : 0.f;
#pragma unroll
      for (int i = 0; i < 8; ++i) {
        const long row = e0 + mi * 32 + mt * 16 + (lh ? (8 + i) : i);
        if (row < E) {
          const float v = acc[mt][nt][i] + bv;
          if (outBf16)
            ((unsigned short*)outp)[(size_t)row * ldo + colg] = f2bf(v);
          else
            ((float*)outp)[(size_t)row * ldo + colg] = v;
        }
      }
    }
  }
}

// ---------------------------------------------------------------------------
extern "C" void kernel_launch(void* const* d_in, const int* in_sizes, int n_in,
                              void* d_out, int out_size, void* d_ws, size_t ws_size,
                              hipStream_t stream) {
  (void)n_in; (void)out_size; (void)ws_size;
  const float* x_emb    = (const float*)d_in[0];
  const float* x_edge   = (const float*)d_in[1];
  const float* rad_w1   = (const float*)d_in[2];
  const float* rad_b1   = (const float*)d_in[3];
  const float* rad_g    = (const float*)d_in[4];
  const float* rad_beta = (const float*)d_in[5];
  const float* rad_w2   = (const float*)d_in[6];
  const float* rad_b2   = (const float*)d_in[7];
  const float* w_m0     = (const float*)d_in[8];
  const float* b_m0     = (const float*)d_in[9];
  const float* w_m1     = (const float*)d_in[10];
  const float* w_m2     = (const float*)d_in[11];
  float* out = (float*)d_out;

  const int E = in_sizes[0] / (29 * 128);  // 30000

  // workspace carve (all sizes multiples of 256B)
  size_t off = 0;
  char* base = (char*)d_ws;
  auto take = [&](size_t bytes) {
    char* p = base + off;
    off += (bytes + 255) & ~(size_t)255;
    return p;
  };
  unsigned short* W2bf  = (unsigned short*)take(2304ull * 128 * 2);
  unsigned short* Wm0bf = (unsigned short*)take(896ull * 896 * 2);
  unsigned short* Wm1bf = (unsigned short*)take(1536ull * 1536 * 2);
  unsigned short* Wm2bf = (unsigned short*)take(1280ull * 1280 * 2);
  unsigned short* h_bf  = (unsigned short*)take((size_t)E * 128 * 2);
  unsigned short* radb  = (unsigned short*)take((size_t)E * 2304 * 2);

  {
    long total = 2304L * 128 + 896L * 896 + 1536L * 1536 + 1280L * 1280;
    int blocks = (int)((total + 255) / 256);
    so2_prep_weights<<<blocks, 256, 0, stream>>>(rad_w2, w_m0, w_m1, w_m2,
                                                 W2bf, Wm0bf, Wm1bf, Wm2bf);
  }
  so2_h_kernel<<<(E + 15) / 16, 256, 0, stream>>>(x_edge, rad_w1, rad_b1, rad_g,
                                                  rad_beta, h_bf, E);
  const int mb = (E + 127) / 128;
  // rad = h @ W2^T + b2  (bf16 out)
  so2_wmma_gemm<<<dim3(18, mb), 256, 0, stream>>>(h_bf, 128, nullptr, nullptr, 0,
                                                  W2bf, rad_b2, (void*)radb, 2304, 1,
                                                  E, 2304, 128);
  // m = 0  (out cols 0..895, with bias)
  so2_wmma_gemm<<<dim3(7, mb), 256, 0, stream>>>(nullptr, 0, x_emb, radb, 0,
                                                 Wm0bf, b_m0, (void*)(out + 0), 3712, 0,
                                                 E, 896, 896);
  // m = 1  (out cols 896..2431, complex recombination folded into Wm1bf)
  so2_wmma_gemm<<<dim3(12, mb), 256, 0, stream>>>(nullptr, 0, x_emb, radb, 1,
                                                  Wm1bf, nullptr, (void*)(out + 896), 3712, 0,
                                                  E, 1536, 1536);
  // m = 2  (out cols 2432..3711)
  so2_wmma_gemm<<<dim3(10, mb), 256, 0, stream>>>(nullptr, 0, x_emb, radb, 2,
                                                  Wm2bf, nullptr, (void*)(out + 2432), 3712, 0,
                                                  E, 1280, 1280);
}
